// DynamicRNN_87522843560198
// MI455X (gfx1250) — matrix operand order repaired and checked by reading, not compiled, not run on hardware
//
#include <hip/hip_runtime.h>
#include <hip/hip_bf16.h>
#include <math.h>

// ---------------- problem constants ----------------
#define BB   32
#define TT   512
#define EE   512
#define HH   1024
#define G3   3072            // 3*H
#define MROWS (BB*TT)        // 16384 rows of x_proj

// ---------------- WMMA types ----------------
typedef __attribute__((ext_vector_type(16))) __bf16 v16bf;
typedef __attribute__((ext_vector_type(8)))  __bf16 v8bf;
typedef __attribute__((ext_vector_type(8)))  float  v8f;

union FragU { v16bf v; v8bf h[2]; };

static __device__ __forceinline__ v8f wmma_bf16(const v16bf& a, const v16bf& b, const v8f& c) {
    return __builtin_amdgcn_wmma_f32_16x16x32_bf16(false, a, false, b, (short)0, c, false, false);
}

// A fragment: 16x32 bf16 tile from row-major source.
// lane<16 : row = lane,    K = k0..k0+15
// lane>=16: row = lane-16, K = k0+16..k0+31
static __device__ __forceinline__ v16bf load_a(const __bf16* rowPtr, int k0, int half) {
    FragU u;
    u.h[0] = *(const v8bf*)(rowPtr + k0 + 16 * half);
    u.h[1] = *(const v8bf*)(rowPtr + k0 + 16 * half + 8);
    return u.v;
}
// B fragment: 32x16 bf16 from transposed ([N,K] row-major) matrix.
// lane<16 : col = lane,    K = k0..k0+15 ; lane>=16: col = lane-16, K = k0+16..k0+31
static __device__ __forceinline__ v16bf load_b(const __bf16* colRow, int k0, int half) {
    FragU u;
    const __bf16* p = colRow + k0 + 16 * half;
    u.h[0] = *(const v8bf*)(p);
    u.h[1] = *(const v8bf*)(p + 8);
    return u.v;
}
// B fragment from LDS staged in fragment order: [chunk(512B)][lane(16B)]
static __device__ __forceinline__ v16bf lds_frag(const char* fragBase, int lane) {
    FragU u;
    u.h[0] = *(const v8bf*)(fragBase + lane * 16);
    u.h[1] = *(const v8bf*)(fragBase + 512 + lane * 16);
    return u.v;
}

static __device__ __forceinline__ float fast_sigmoid(float x) {
    return 1.0f / (1.0f + __expf(-x));
}
static __device__ __forceinline__ float fast_tanh(float x) {
    return 1.0f - 2.0f / (__expf(2.0f * x) + 1.0f);
}

// ---------------- phase 0: conversions ----------------
__global__ void k_f32_to_bf16(const float* __restrict__ in, __bf16* __restrict__ out, int n) {
    int i = blockIdx.x * blockDim.x + threadIdx.x;
    if (i < n) out[i] = (__bf16)in[i];
}

// in: [rows_in, cols_in] f32 -> out: [cols_in, rows_in] bf16 (transpose + convert)
__global__ void k_transpose_bf16(const float* __restrict__ in, __bf16* __restrict__ out,
                                 int rows_in, int cols_in) {
    int i = blockIdx.x * blockDim.x + threadIdx.x;
    int n = rows_in * cols_in;
    if (i < n) {
        int r = i / rows_in;       // output row == input col
        int c = i - r * rows_in;   // output col == input row
        out[i] = (__bf16)in[c * cols_in + r];
    }
}

// ---------------- phase 1: x_proj = X @ W + b_in ----------------
// A: [16384, 512] bf16 row-major; Wt: [3072, 512] bf16; out xp: [16384,3072] f32.
// One wave = 32 rows (2 M-tiles) x 64 cols (4 N-tiles), K=512.
// Fully unrolled k-loop, distance-1 rotating fragment buffers (renamed regs).
__global__ __launch_bounds__(128, 1) void k_xproj_gemm(const __bf16* __restrict__ A,
                                                       const __bf16* __restrict__ Wt,
                                                       const float* __restrict__ b_in,
                                                       float* __restrict__ xp) {
    const int lane = threadIdx.x & 31;
    const int half = lane >> 4;
    const int wid  = (blockIdx.x * blockDim.x + threadIdx.x) >> 5; // 0..24575
    const int mt   = wid / 48;   // 0..511  pair of M-tiles
    const int ng   = wid % 48;   // 0..47   group of 4 N-tiles
    const int m0   = mt * 32;
    const int n0   = ng * 64;

    v8f acc[2][4];
#pragma unroll
    for (int a2 = 0; a2 < 2; ++a2)
#pragma unroll
        for (int nt = 0; nt < 4; ++nt) acc[a2][nt] = (v8f){};

    const __bf16* aRow0 = A + (size_t)(m0 + (lane & 15)) * EE;
    const __bf16* aRow1 = A + (size_t)(m0 + 16 + (lane & 15)) * EE;
    const __bf16* bRow[4];
#pragma unroll
    for (int nt = 0; nt < 4; ++nt)
        bRow[nt] = Wt + (size_t)(n0 + nt * 16 + (lane & 15)) * EE;

    // distance-1 rotating buffers (full unroll -> pure register renaming)
    v16bf a0p[2], a1p[2], bp[2][4];
    a0p[0] = load_a(aRow0, 0, half);
    a1p[0] = load_a(aRow1, 0, half);
#pragma unroll
    for (int nt = 0; nt < 4; ++nt) bp[0][nt] = load_b(bRow[nt], 0, half);

#pragma unroll
    for (int s = 0; s < 16; ++s) {
        const int cur = s & 1, nxt = (s + 1) & 1;
        if (s < 15) {
            a0p[nxt] = load_a(aRow0, (s + 1) * 32, half);
            a1p[nxt] = load_a(aRow1, (s + 1) * 32, half);
#pragma unroll
            for (int nt = 0; nt < 4; ++nt) bp[nxt][nt] = load_b(bRow[nt], (s + 1) * 32, half);
        }
#pragma unroll
        for (int nt = 0; nt < 4; ++nt) acc[0][nt] = wmma_bf16(a0p[cur], bp[cur][nt], acc[0][nt]);
#pragma unroll
        for (int nt = 0; nt < 4; ++nt) acc[1][nt] = wmma_bf16(a1p[cur], bp[cur][nt], acc[1][nt]);
    }

#pragma unroll
    for (int a2 = 0; a2 < 2; ++a2) {
#pragma unroll
        for (int nt = 0; nt < 4; ++nt) {
            int n = n0 + nt * 16 + (lane & 15);
            float bi = b_in[n];
#pragma unroll
            for (int v = 0; v < 8; ++v) {
                int m = m0 + a2 * 16 + 8 * half + v;
                xp[(size_t)m * G3 + n] = acc[a2][nt][v] + bi;
            }
        }
    }
}

// ---------------- phase 2: persistent GRU scan ----------------
// 32 blocks x 64 threads = 64 waves. Wave j owns columns [16j,16j+16) of H and
// both M-tiles (batch rows 0-15 / 16-31) of the z/r/h gate triple.
// B operands (time-invariant 96KB/wave of U^T) staged once in LDS in fragment
// order; only h (64KB bf16) streams from L2 each step (distance-2 pipeline).
#define SCAN_BLOCKS 32
#define SCAN_THREADS 64
#define WAVE_LDS (3 * 32 * 1024)     // 96KB: 3 gates x 32 k-steps x 1KB frag
#define SCAN_LDS (2 * WAVE_LDS)      // 192KB per block (<= 320KB WGP LDS)

__global__ __launch_bounds__(SCAN_THREADS, 1) void k_gru_scan(
        const float* __restrict__ xp, const __bf16* __restrict__ Ut,
        const float* __restrict__ brec, float* __restrict__ y,
        float* __restrict__ state, __bf16* __restrict__ hping,
        unsigned* __restrict__ bar) {
    extern __shared__ char smem[];
    const int lane = threadIdx.x & 31;
    const int half = lane >> 4;
    const int wib  = threadIdx.x >> 5;                 // wave in block: 0..1
    const int w    = blockIdx.x * 2 + wib;             // 0..63
    const int n    = w * 16 + (lane & 15);             // gate column, 0..1023
    const unsigned nblk = gridDim.x;

    const float bz = brec[n];
    const float br = brec[HH + n];
    const float bh = brec[2 * HH + n];

    // ---- stage this wave's B fragments (U^T columns) into LDS, frag order ----
    char* ldsw = smem + wib * WAVE_LDS;
    for (int s = 0; s < 32; ++s) {
#pragma unroll
        for (int g = 0; g < 3; ++g) {
            const __bf16* src = Ut + (size_t)(g * HH + n) * HH + s * 32 + 16 * half;
            v8bf c0 = *(const v8bf*)(src);
            v8bf c1 = *(const v8bf*)(src + 8);
            char* dst = ldsw + (s * 3 + g) * 1024 + lane * 16;
            *(v8bf*)(dst)       = c0;
            *(v8bf*)(dst + 512) = c1;
        }
    }
    __syncthreads();

    __bf16* hb0 = hping;
    __bf16* hb1 = hping + BB * HH;
    float hreg[2][8] = {{0.f}};   // this lane's owned h elements (f32)

    for (int t = 0; t < TT; ++t) {
        const __bf16* hr = (t & 1) ? hb1 : hb0;   // read buffer (prev h, bf16)
        __bf16*       hw = (t & 1) ? hb0 : hb1;   // write buffer (new h)

        const __bf16* a0row = hr + (size_t)(lane & 15) * HH;        // rows 0..15
        const __bf16* a1row = hr + (size_t)(16 + (lane & 15)) * HH; // rows 16..31

        // ---- A (h) prologue FIRST so first-WMMA waits don't cover xg loads ----
        v16bf a0p[3], a1p[3];
        a0p[0] = load_a(a0row, 0, half);
        a1p[0] = load_a(a1row, 0, half);
        a0p[1] = load_a(a0row, 32, half);
        a1p[1] = load_a(a1row, 32, half);

        // prefetch this step's x_proj gate inputs (hidden behind the GEMM)
        float xg[3][2][8];
#pragma unroll
        for (int mtv = 0; mtv < 2; ++mtv) {
#pragma unroll
            for (int v = 0; v < 8; ++v) {
                int m = mtv * 16 + 8 * half + v;
                size_t rb = ((size_t)m * TT + t) * (size_t)G3;
                xg[0][mtv][v] = xp[rb + n];
                xg[1][mtv][v] = xp[rb + HH + n];
                xg[2][mtv][v] = xp[rb + 2 * HH + n];
            }
        }

        v8f acc[2][3];
#pragma unroll
        for (int mt2 = 0; mt2 < 2; ++mt2)
#pragma unroll
            for (int g = 0; g < 3; ++g) acc[mt2][g] = (v8f){};

        // LDS B prologue (distance-1)
        v16bf fp[2][3];
#pragma unroll
        for (int g = 0; g < 3; ++g) fp[0][g] = lds_frag(ldsw + g * 1024, lane);

        // fully unrolled k-loop: A ring distance 2, LDS ring distance 1
#pragma unroll
        for (int s = 0; s < 32; ++s) {
            const int ac = s % 3, an = (s + 2) % 3;
            const int fc = s & 1, fn = (s + 1) & 1;
            if (s < 30) {
                a0p[an] = load_a(a0row, (s + 2) * 32, half);
                a1p[an] = load_a(a1row, (s + 2) * 32, half);
            }
            if (s < 31) {
                const char* fb = ldsw + ((s + 1) * 3) * 1024;
#pragma unroll
                for (int g = 0; g < 3; ++g) fp[fn][g] = lds_frag(fb + g * 1024, lane);
            }
            acc[0][0] = wmma_bf16(a0p[ac], fp[fc][0], acc[0][0]);
            acc[1][0] = wmma_bf16(a1p[ac], fp[fc][0], acc[1][0]);
            acc[0][1] = wmma_bf16(a0p[ac], fp[fc][1], acc[0][1]);
            acc[1][1] = wmma_bf16(a1p[ac], fp[fc][1], acc[1][1]);
            acc[0][2] = wmma_bf16(a0p[ac], fp[fc][2], acc[0][2]);
            acc[1][2] = wmma_bf16(a1p[ac], fp[fc][2], acc[1][2]);
        }

#pragma unroll
        for (int mt2 = 0; mt2 < 2; ++mt2) {
#pragma unroll
            for (int v = 0; v < 8; ++v) {
                int m = mt2 * 16 + 8 * half + v;                 // batch row
                float rz = acc[mt2][0][v] + bz;
                float rr = acc[mt2][1][v] + br;
                float rh = acc[mt2][2][v] + bh;
                float z  = fast_sigmoid(xg[0][mt2][v] + rz);
                float r  = fast_sigmoid(xg[1][mt2][v] + rr);
                float hh = fast_tanh(xg[2][mt2][v] + r * rh);
                float hn = z * hreg[mt2][v] + (1.0f - z) * hh;
                hreg[mt2][v] = hn;
                y[((size_t)m * TT + t) * (size_t)HH + n] = hn;
                hw[(size_t)m * HH + n] = (__bf16)hn;
                if (t == TT - 1) state[(size_t)m * HH + n] = hn;
            }
        }

        // ---- grid barrier: release h writes, acquire (per-CU inv) in every wave ----
        __threadfence();
        __syncthreads();
        if (threadIdx.x == 0) {
            __hip_atomic_fetch_add(bar, 1u, __ATOMIC_RELEASE, __HIP_MEMORY_SCOPE_AGENT);
        }
        if (lane == 0) {   // acquire-spin in EVERY wave: invalidates each CU's L0
            const unsigned target = (unsigned)(t + 1) * nblk;
            while (__hip_atomic_load(bar, __ATOMIC_ACQUIRE, __HIP_MEMORY_SCOPE_AGENT) < target) {
                __builtin_amdgcn_s_sleep(1);
            }
        }
        __syncthreads();
    }
}

// ---------------- workspace layout (bytes, all 256-aligned) ----------------
#define SZ_A   ((size_t)MROWS * EE * 2)          //  16 MB  input bf16
#define SZ_WT  ((size_t)G3 * EE * 2)             //   3 MB  W^T bf16
#define SZ_UT  ((size_t)G3 * HH * 2)             //   6 MB  U^T bf16
#define SZ_XP  ((size_t)MROWS * G3 * 4)          // 192 MB  x_proj f32
#define SZ_H   ((size_t)2 * BB * HH * 2)         // 128 KB  h ping-pong bf16
#define SZ_BAR ((size_t)256)

extern "C" void kernel_launch(void* const* d_in, const int* in_sizes, int n_in,
                              void* d_out, int out_size, void* d_ws, size_t ws_size,
                              hipStream_t stream) {
    const float* inp = (const float*)d_in[0];   // [32,512,512]
    const float* W   = (const float*)d_in[1];   // [512,3072]
    const float* U   = (const float*)d_in[2];   // [1024,3072]
    const float* b   = (const float*)d_in[3];   // [2,3072]
    float* y = (float*)d_out;                   // [32,512,1024] ++ [32,1024]
    float* state = y + (size_t)BB * TT * HH;

    char* ws = (char*)d_ws;
    size_t offA  = 0;
    size_t offWt = offA + SZ_A;
    size_t offUt = offWt + SZ_WT;
    size_t offXp = offUt + SZ_UT;
    size_t offH  = offXp + SZ_XP;
    size_t offB  = offH + SZ_H;
    size_t total = offB + SZ_BAR;
    if (ws_size < total) return;   // ~228 MB scratch required

    __bf16* Abf = (__bf16*)(ws + offA);
    __bf16* Wt  = (__bf16*)(ws + offWt);
    __bf16* Ut  = (__bf16*)(ws + offUt);
    float*  xp  = (float*)(ws + offXp);
    __bf16* hp  = (__bf16*)(ws + offH);
    unsigned* bar = (unsigned*)(ws + offB);

    // zero h ping-pong + barrier counter (capturable async memset)
    hipMemsetAsync(ws + offH, 0, SZ_H + SZ_BAR, stream);

    // phase 0: convert / transpose weights
    {
        int nels = MROWS * EE;                               // 8,388,608
        k_f32_to_bf16<<<(nels + 255) / 256, 256, 0, stream>>>(inp, Abf, nels);
    }
    {
        int nels = EE * G3;                                  // 1,572,864
        k_transpose_bf16<<<(nels + 255) / 256, 256, 0, stream>>>(W, Wt, EE, G3);
    }
    {
        int nels = HH * G3;                                  // 3,145,728
        k_transpose_bf16<<<(nels + 255) / 256, 256, 0, stream>>>(U, Ut, HH, G3);
    }

    // phase 1: x_proj GEMM (512 M-pairs x 48 groups = 24576 waves, 4 waves/block)
    k_xproj_gemm<<<6144, 128, 0, stream>>>(Abf, Wt, b, xp);

    // phase 2: persistent recurrent scan, 192KB dynamic LDS per block
    k_gru_scan<<<SCAN_BLOCKS, SCAN_THREADS, SCAN_LDS, stream>>>(
        xp, Ut, b + G3, y, state, hp, bar);
}
